// HyperbolicAlignmentLoss_85804856640309
// MI455X (gfx1250) — matrix operand order repaired and checked by reading, not compiled, not run on hardware
//
#include <hip/hip_runtime.h>
#include <hip/hip_bf16.h>
#include <math.h>

typedef __attribute__((ext_vector_type(16))) __bf16 v16bf;
typedef __attribute__((ext_vector_type(8)))  float  v8f;
typedef __attribute__((ext_vector_type(4)))  unsigned int v4u;
typedef __attribute__((ext_vector_type(8)))  int v8i_;
typedef __attribute__((ext_vector_type(4)))  int v4i_;

#define BDIM 8192
#define DDIM 1024
#define MBLK 64
#define NBLK 128
#define KCH  32
#define ASTR 40            // padded LDS row stride (bf16): 80B = 20 dwords -> conflict-free b128
#define NSPLIT 8
#define COLCHUNK (BDIM / NSPLIT)   // 1024 columns per workgroup
#define LSMOOTH 0.1f
#define EPSV 1e-7f

// ---------------------------------------------------------------------------
// Kernel 0: zero per-row accumulators
// ---------------------------------------------------------------------------
__global__ void zero_kernel(float* __restrict__ p, int n) {
    int i = blockIdx.x * blockDim.x + threadIdx.x;
    if (i < n) p[i] = 0.f;
}

// ---------------------------------------------------------------------------
// Kernel 1: fp32 -> bf16 hi/lo split + row squared norms (exact fp32)
// ---------------------------------------------------------------------------
__global__ void __launch_bounds__(256)
prep_kernel(const float* __restrict__ sign, const float* __restrict__ text,
            __bf16* __restrict__ signHi, __bf16* __restrict__ signLo,
            __bf16* __restrict__ textHi, __bf16* __restrict__ textLo,
            float* __restrict__ xx, float* __restrict__ yy)
{
    int b = blockIdx.x;
    const float* src; __bf16 *hi, *lo; float* nrm; int row;
    if (b < BDIM) { src = sign; hi = signHi; lo = signLo; nrm = xx; row = b; }
    else          { src = text; hi = textHi; lo = textLo; nrm = yy; row = b - BDIM; }

    int tid = threadIdx.x;
    size_t base = (size_t)row * DDIM;
    float acc = 0.f;
    for (int e = tid; e < DDIM; e += 256) {
        float x = src[base + e];
        __bf16 h = (__bf16)x;
        float hf = (float)h;
        hi[base + e] = h;
        lo[base + e] = (__bf16)(x - hf);
        acc += x * x;
    }
    __shared__ float red[256];
    red[tid] = acc; __syncthreads();
    for (int s = 128; s > 0; s >>= 1) {
        if (tid < s) red[tid] += red[tid + s];
        __syncthreads();
    }
    if (tid == 0) nrm[row] = red[0];
}

// ---------------------------------------------------------------------------
// Tensor Data Mover: async DMA of a 2D bf16 tile (tileRows x KCH) into LDS
// with hardware row padding (64B data + 16B pad -> LDS stride = ASTR bf16).
// Descriptor per CDNA5 ISA ch.8; 6-arg clang-23 builtin. Tracked by TENSORcnt.
// ---------------------------------------------------------------------------
__device__ __forceinline__ unsigned lds_off(const void* p) {
    return (unsigned)(size_t)p;     // low 32 bits of generic addr = LDS offset
}

__device__ __forceinline__ void tdm_load_tile(unsigned ldsAddr,
                                              const __bf16* gptr,
                                              int tileRows)
{
    unsigned long long ga = (unsigned long long)(size_t)gptr;
    const unsigned td0 = 0x100000u;          // generous dims: tiles never OOB
    const unsigned td1 = 0x100000u;

    v4u g0;
    g0[0] = 1u;                                           // count=1, user mode
    g0[1] = ldsAddr;                                      // LDS byte address
    g0[2] = (unsigned)(ga & 0xffffffffu);                 // global addr [31:0]
    g0[3] = (unsigned)((ga >> 32) & 0x1ffffffu)           // global addr [56:32]
          | (2u << 30);                                   // type = 2 (image)

    v8i_ g1;
    // data_size=1 (2B) | pad_enable | pad_interval=3 (16 dw = 64B) | pad_amount=3 (4 dw = 16B)
    g1[0] = (int)((1u << 16) | (1u << 20) | (3u << 22) | (3u << 25));
    g1[1] = (int)((td0 & 0xffffu) << 16);                 // tensor_dim0 lo
    g1[2] = (int)(((td0 >> 16) & 0xffffu) | ((td1 & 0xffffu) << 16));
    g1[3] = (int)(((td1 >> 16) & 0xffffu) | ((unsigned)KCH << 16));  // tile_dim0 = 32
    g1[4] = (int)((unsigned)tileRows & 0xffffu);          // tile_dim1, tile_dim2=0
    g1[5] = DDIM;                                         // tensor_dim0_stride lo (elems)
    g1[6] = 0;
    g1[7] = 0;

    v4i_ z4 = {0, 0, 0, 0};                               // 2D tensor: groups 2/3 unused
    v8i_ z8 = {0, 0, 0, 0, 0, 0, 0, 0};
    __builtin_amdgcn_tensor_load_to_lds(g0, g1, z4, z4, z8, 0);
}

// ---------------------------------------------------------------------------
// LDS fragment loaders (layouts per CDNA5 ISA 7.12.2, wave32)
// ---------------------------------------------------------------------------
__device__ __forceinline__ v16bf loadA(const __bf16* sh, int row, int half) {
    union { uint4 q[2]; v16bf v; } f;
    const __bf16* p = sh + row * ASTR + half * 8;
    f.q[0] = *(const uint4*)p;
    f.q[1] = *(const uint4*)(p + 16);
    return f.v;
}
__device__ __forceinline__ v16bf loadB(const __bf16* sh, int col, int half) {
    union { uint4 q[2]; v16bf v; } f;
    const __bf16* p = sh + col * ASTR + half * 8;
    f.q[0] = *(const uint4*)p;
    f.q[1] = *(const uint4*)(p + 16);
    return f.v;
}

// ---------------------------------------------------------------------------
// Kernel 2: TDM double-buffered fused Gram WMMA (bf16 hi/lo split, 3 WMMAs)
// + Poincare distance + fixed-shift softmax row stats.
// Grid (B/MBLK, NSPLIT), 256 threads (8 waves: 2 Mx4 N).
// ---------------------------------------------------------------------------
__global__ void __launch_bounds__(256)
gram_fused_kernel(const __bf16* __restrict__ signHi, const __bf16* __restrict__ signLo,
                  const __bf16* __restrict__ textHi, const __bf16* __restrict__ textLo,
                  const float* __restrict__ xx, const float* __restrict__ yy,
                  const float* __restrict__ logTauP, const float* __restrict__ marginP,
                  float* __restrict__ rowS, float* __restrict__ rowT,
                  float* __restrict__ dD, float* __restrict__ dS)
{
    __shared__ __align__(16) __bf16 shA[2][2][MBLK * ASTR];   // [buf][hi|lo] 20.0 KB
    __shared__ __align__(16) __bf16 shB[2][2][NBLK * ASTR];   // [buf][hi|lo] 40.0 KB
    __shared__ float s_xx[MBLK], s_rowS[MBLK], s_rowT[MBLK], s_dD[MBLK], s_dS[MBLK];

    const int tid  = threadIdx.x;
    const int lane = tid & 31;
    const int wv   = tid >> 5;
    const int wm   = wv & 1;          // wave row (0..1) -> 32 rows each
    const int wn   = wv >> 1;         // wave col (0..3) -> 32 cols each
    const int half = lane >> 4;
    const int l16  = lane & 15;

    const int rowBase = blockIdx.x * MBLK;

    const float lt  = logTauP[0];
    const float tau = 1.99f / (1.f + __expf(-lt)) + 0.01f;
    const float invTau = 1.f / tau;
    const float mEff = fmaxf(marginP[0], 0.f);   // softmax shift (upper bound of scores)

    if (tid < MBLK) {
        s_xx[tid] = xx[rowBase + tid];
        s_rowS[tid] = 0.f; s_rowT[tid] = 0.f;
    }
    const bool diagBlock = ((int)blockIdx.y == (int)(blockIdx.x >> 4));  // COLCHUNK/MBLK = 16

    for (int pass = 0; pass < COLCHUNK / NBLK; ++pass) {
        const int colBase = blockIdx.y * COLCHUNK + pass * NBLK;
        v8f acc[2][2] = {};
        int buf = 0;

        // prologue: kick DMA for first K-chunk (waves 0-3, one tile stream each)
        {
            size_t gA = (size_t)rowBase * DDIM;
            size_t gB = (size_t)colBase * DDIM;
            if      (wv == 0) tdm_load_tile(lds_off(&shA[0][0][0]), signHi + gA, MBLK);
            else if (wv == 1) tdm_load_tile(lds_off(&shA[0][1][0]), signLo + gA, MBLK);
            else if (wv == 2) tdm_load_tile(lds_off(&shB[0][0][0]), textHi + gB, NBLK);
            else if (wv == 3) tdm_load_tile(lds_off(&shB[0][1][0]), textLo + gB, NBLK);
        }

        for (int kb = 0; kb < DDIM; kb += KCH) {
            if (wv < 4) __builtin_amdgcn_s_wait_tensorcnt((short)0);  // current buf ready
            __syncthreads();   // all waves done with buf^1 from previous chunk + see DMA data

            if (kb + KCH < DDIM) {   // overlap next chunk's DMA with this chunk's WMMAs
                size_t gA = (size_t)rowBase * DDIM + kb + KCH;
                size_t gB = (size_t)colBase * DDIM + kb + KCH;
                int nb = buf ^ 1;
                if      (wv == 0) tdm_load_tile(lds_off(&shA[nb][0][0]), signHi + gA, MBLK);
                else if (wv == 1) tdm_load_tile(lds_off(&shA[nb][1][0]), signLo + gA, MBLK);
                else if (wv == 2) tdm_load_tile(lds_off(&shB[nb][0][0]), textHi + gB, NBLK);
                else if (wv == 3) tdm_load_tile(lds_off(&shB[nb][1][0]), textLo + gB, NBLK);
            }

            const __bf16* aHi = shA[buf][0];
            const __bf16* aLo = shA[buf][1];
            const __bf16* bHi = shB[buf][0];
            const __bf16* bLo = shB[buf][1];

            v16bf aH[2], aL[2], bH[2], bL[2];
            #pragma unroll
            for (int mi = 0; mi < 2; ++mi) {
                int r = wm * 32 + mi * 16 + l16;
                aH[mi] = loadA(aHi, r, half);
                aL[mi] = loadA(aLo, r, half);
            }
            #pragma unroll
            for (int ni = 0; ni < 2; ++ni) {
                int c = wn * 32 + ni * 16 + l16;
                bH[ni] = loadB(bHi, c, half);
                bL[ni] = loadB(bLo, c, half);
            }
            #pragma unroll
            for (int mi = 0; mi < 2; ++mi)
                #pragma unroll
                for (int ni = 0; ni < 2; ++ni) {
                    acc[mi][ni] = __builtin_amdgcn_wmma_f32_16x16x32_bf16(
                        false, aH[mi], false, bH[ni], (short)0, acc[mi][ni], false, false);
                    acc[mi][ni] = __builtin_amdgcn_wmma_f32_16x16x32_bf16(
                        false, aH[mi], false, bL[ni], (short)0, acc[mi][ni], false, false);
                    acc[mi][ni] = __builtin_amdgcn_wmma_f32_16x16x32_bf16(
                        false, aL[mi], false, bH[ni], (short)0, acc[mi][ni], false, false);
                }
            buf ^= 1;
        }

        // epilogue: Poincare dist + score + fixed-shift softmax stats.
        // C layout: VGPR v, lanes 0-15 -> M=v (N=l16); lanes 16-31 -> M=v+8.
        #pragma unroll
        for (int mi = 0; mi < 2; ++mi) {
            #pragma unroll
            for (int v = 0; v < 8; ++v) {
                int rl = wm * 32 + mi * 16 + v + half * 8;
                int gRow = rowBase + rl;
                float xr  = s_xx[rl];
                float omx = 1.f - xr;
                float eAcc = 0.f, tAcc = 0.f;
                #pragma unroll
                for (int ni = 0; ni < 2; ++ni) {
                    int col  = colBase + wn * 32 + ni * 16 + l16;
                    float xy = acc[mi][ni][v];
                    float yc = yy[col];
                    float sq  = fmaxf(xr + yc - 2.f * xy, 0.f);
                    float den = fmaxf(omx * (1.f - yc), EPSV);
                    float arg = fmaxf(1.f + 2.f * sq / den, 1.f + EPSV);
                    float d   = __logf(arg + __fsqrt_rn(arg * arg - 1.f));   // arccosh
                    bool diag = (col == gRow);
                    float s   = -d * invTau + (diag ? 0.f : mEff);
                    if (diag) { s_dD[rl] = d; s_dS[rl] = s; }
                    eAcc += __expf(s - mEff);     // scores <= mEff, no overflow
                    tAcc += s;
                }
                #pragma unroll
                for (int off = 1; off < 16; off <<= 1) {   // 16-lane row reduction
                    eAcc += __shfl_xor(eAcc, off, 32);
                    tAcc += __shfl_xor(tAcc, off, 32);
                }
                if (l16 == 0) {
                    atomicAdd(&s_rowS[rl], eAcc);
                    atomicAdd(&s_rowT[rl], tAcc);
                }
            }
        }
    }

    __syncthreads();
    if (tid < MBLK) {
        int r = rowBase + tid;
        atomicAdd(&rowS[r], s_rowS[tid]);
        atomicAdd(&rowT[r], s_rowT[tid]);
        if (diagBlock) { dD[r] = s_dD[tid]; dS[r] = s_dS[tid]; }
    }
}

// ---------------------------------------------------------------------------
// Kernel 3: fold row statistics into the 4 scalar outputs
// ---------------------------------------------------------------------------
__global__ void __launch_bounds__(256)
final_kernel(const float* __restrict__ rowS, const float* __restrict__ rowT,
             const float* __restrict__ dD, const float* __restrict__ dS,
             const float* __restrict__ logTauP, const float* __restrict__ marginP,
             float* __restrict__ out)
{
    int tid = threadIdx.x;
    float sL = 0.f, sT = 0.f, sDS = 0.f, sDD = 0.f;
    for (int i = tid; i < BDIM; i += 256) {
        sL  += __logf(rowS[i]);
        sT  += rowT[i];
        sDS += dS[i];
        sDD += dD[i];
    }
    __shared__ float r0[256], r1[256], r2[256], r3[256];
    r0[tid] = sL; r1[tid] = sT; r2[tid] = sDS; r3[tid] = sDD;
    __syncthreads();
    for (int s = 128; s > 0; s >>= 1) {
        if (tid < s) { r0[tid] += r0[tid+s]; r1[tid] += r1[tid+s];
                       r2[tid] += r2[tid+s]; r3[tid] += r3[tid+s]; }
        __syncthreads();
    }
    if (tid == 0) {
        float lt  = logTauP[0];
        float tau = 1.99f / (1.f + __expf(-lt)) + 0.01f;
        float mg  = marginP[0];
        float mEff = fmaxf(mg, 0.f);
        float invB = 1.f / (float)BDIM;
        float meanLse = mEff + r0[0] * invB;
        float loss = meanLse - (1.f - LSMOOTH) * (r2[0] * invB)
                             - LSMOOTH * (r1[0] * invB * invB);
        out[0] = loss;
        out[1] = r3[0] * invB;   // pos_dist
        out[2] = tau;
        out[3] = mg;             // jnp.maximum(margin, margin) == margin
    }
}

// ---------------------------------------------------------------------------
extern "C" void kernel_launch(void* const* d_in, const int* in_sizes, int n_in,
                              void* d_out, int out_size, void* d_ws, size_t ws_size,
                              hipStream_t stream)
{
    const float* sign   = (const float*)d_in[0];
    const float* text   = (const float*)d_in[1];
    const float* logTau = (const float*)d_in[2];
    const float* margin = (const float*)d_in[3];
    float* out = (float*)d_out;

    char* ws = (char*)d_ws;
    size_t off = 0;
    auto alloc = [&](size_t bytes) -> void* {
        void* p = ws + off;
        off += (bytes + 255) & ~(size_t)255;
        return p;
    };
    __bf16* signHi = (__bf16*)alloc((size_t)BDIM * DDIM * 2);
    __bf16* signLo = (__bf16*)alloc((size_t)BDIM * DDIM * 2);
    __bf16* textHi = (__bf16*)alloc((size_t)BDIM * DDIM * 2);
    __bf16* textLo = (__bf16*)alloc((size_t)BDIM * DDIM * 2);
    float* xx   = (float*)alloc((size_t)BDIM * 4);
    float* yy   = (float*)alloc((size_t)BDIM * 4);
    float* rowS = (float*)alloc((size_t)BDIM * 4);
    float* rowT = (float*)alloc((size_t)BDIM * 4);
    float* dD   = (float*)alloc((size_t)BDIM * 4);
    float* dS   = (float*)alloc((size_t)BDIM * 4);
    (void)ws_size; (void)in_sizes; (void)n_in; (void)out_size;

    zero_kernel<<<(BDIM + 255) / 256, 256, 0, stream>>>(rowS, BDIM);
    zero_kernel<<<(BDIM + 255) / 256, 256, 0, stream>>>(rowT, BDIM);

    prep_kernel<<<2 * BDIM, 256, 0, stream>>>(sign, text, signHi, signLo,
                                              textHi, textLo, xx, yy);

    gram_fused_kernel<<<dim3(BDIM / MBLK, NSPLIT), 256, 0, stream>>>(
        signHi, signLo, textHi, textLo, xx, yy, logTau, margin,
        rowS, rowT, dD, dS);

    final_kernel<<<1, 256, 0, stream>>>(rowS, rowT, dD, dS, logTau, margin, out);
}
